// NUFNO1D_61529701482700
// MI455X (gfx1250) — matrix operand order repaired and checked by reading, not compiled, not run on hardware
//
#include <hip/hip_runtime.h>
#include <stdint.h>

typedef __attribute__((ext_vector_type(16))) __bf16 v16bf;
typedef __attribute__((ext_vector_type(8)))  float  v8f;

#define BB      32
#define NTOT    131072
#define NSUB    16
#define MSUB    8192
#define SG      1024
#define AL      2048
#define MODES_  32
#define W_      64
#define NLAYERS 4
#define GRID_TOT (NSUB*SG)
#define EPSF 1e-6f
#define PI_F 3.14159265358979323846f

// GEMM epilogue / staging flags
#define F_GELU_B 1
#define F_GELU_A 2
#define F_ACCUM  4
#define F_BIAS   8
#define F_RES    16

__device__ __forceinline__ float gelu_f(float x) {
  float x3 = x * x * x;
  return 0.5f * x * (1.0f + tanhf(0.79788456080286535588f * (x + 0.044715f * x3)));
}

__device__ __forceinline__ unsigned int pack_bf16(float lo, float hi) {
  union { __bf16 h[2]; unsigned int u; } p;
  p.h[0] = (__bf16)lo; p.h[1] = (__bf16)hi;
  return p.u;
}

// ---------------------------------------------------------------------------
// Generic bf16-WMMA GEMM:  C[M,N] = op(A)[M,K] * op(B)[K,N]  (+bias +res +C)
// A,B are f32 in global. Tiles are staged into LDS **in WMMA fragment order**
// (inverse of ISA 7.12.2 16-bit A 16x32 / B 32x16 lane maps), bf16-pair packed,
// so each fragment is consumed as one 32-byte LDS read (2x ds_load_b128).
// One 16-row m-tile per wave, 64-col n-block per workgroup, K stepped by 32.
// ---------------------------------------------------------------------------
__global__ void gemm_wmma_bf16(const float* __restrict__ A, const float* __restrict__ B,
                               float* __restrict__ C, const float* __restrict__ bias,
                               const float* __restrict__ Res,
                               int M, int N, int K, int lda, int ldb, int ldc,
                               long long sA, long long sB, long long sC, long long sRes,
                               int flags)
{
  // fragment-order buffers: [tile][lane][8 uints] ; 32B per lane -> conflict-free
  __shared__ __attribute__((aligned(32))) unsigned int ldsAf[4 * 32 * 8];
  __shared__ __attribute__((aligned(32))) unsigned int ldsBf[4 * 32 * 8];

  const int mtiles = (int)(blockDim.x >> 5);
  const int tid   = threadIdx.x;
  const int lane  = tid & 31;
  const int wave  = tid >> 5;
  const int lhalf = lane >> 4;
  const int lm    = lane & 15;
  const int batch = blockIdx.y;
  const int n0    = blockIdx.x * 64;
  const int row0  = blockIdx.z * (mtiles * 16);

  const float* Ab = A + (long long)batch * sA;
  const float* Bb = B + (long long)batch * sB;
  float*       Cb = C + (long long)batch * sC;
  const float* Rb = (flags & F_RES) ? (Res + (long long)batch * sRes) : (const float*)0;

  v8f acc[4];
  acc[0] = (v8f){}; acc[1] = (v8f){}; acc[2] = (v8f){}; acc[3] = (v8f){};

  const int aSlots = mtiles * 32 * 8;   // uint slots (2 bf16 each)
  const int bSlots = 4 * 32 * 8;

  for (int k0 = 0; k0 < K; k0 += 32) {
    // prefetch next k-step tiles into cache (global_prefetch_b8)
    if (k0 + 32 < K) {
      __builtin_prefetch(&Bb[(long long)(k0 + 32) * ldb + n0], 0, 1);
      __builtin_prefetch(&Ab[(long long)row0 * lda + (k0 + 32)], 0, 1);
    }

    // ---- stage A in fragment order: lane=(m&15)+16*((k>>3)&1), e=(k&7)+8*(k>>4)
    for (int idx = tid; idx < aSlots; idx += blockDim.x) {
      int mt     = idx >> 8;            // / 256
      int lane_s = (idx >> 3) & 31;
      int e0     = (idx & 7) * 2;
      int m_loc  = lane_s & 15;
      int lh     = lane_s >> 4;
      int k      = (e0 & 7) + (lh << 3) + ((e0 >> 3) << 4);   // k(e0+1)=k+1
      int gr = row0 + mt * 16 + m_loc;
      int gk = k0 + k;
      float v0 = 0.0f, v1 = 0.0f;
      if (gr < M) {
        const float* ap = Ab + (long long)gr * lda;
        if (gk     < K) v0 = ap[gk];
        if (gk + 1 < K) v1 = ap[gk + 1];
      }
      if (flags & F_GELU_A) { v0 = gelu_f(v0); v1 = gelu_f(v1); }
      ldsAf[idx] = pack_bf16(v0, v1);
    }
    // ---- stage B in fragment order: lane=n+16*((k>>3)&1), e=(k&7)+8*(k>>4)
    for (int idx = tid; idx < bSlots; idx += blockDim.x) {
      int ns     = idx >> 8;
      int lane_s = (idx >> 3) & 31;
      int e0     = (idx & 7) * 2;
      int n_loc  = lane_s & 15;
      int kh     = lane_s >> 4;
      int k      = (e0 & 7) + (kh << 3) + ((e0 >> 3) << 4);
      int gk = k0 + k;
      int gc = n0 + ns * 16 + n_loc;
      float v0 = 0.0f, v1 = 0.0f;
      if (gc < N) {
        if (gk     < K) v0 = Bb[(long long)gk * ldb + gc];
        if (gk + 1 < K) v1 = Bb[(long long)(gk + 1) * ldb + gc];
      }
      if (flags & F_GELU_B) { v0 = gelu_f(v0); v1 = gelu_f(v1); }
      ldsBf[idx] = pack_bf16(v0, v1);
    }
    __syncthreads();

    // ---- consume fragments: one 32B LDS read each (2x ds_load_b128)
    v16bf afr = *(const v16bf*)&ldsAf[(wave * 32 + lane) * 8];
#pragma unroll
    for (int ns = 0; ns < 4; ++ns) {
      v16bf bfr = *(const v16bf*)&ldsBf[(ns * 32 + lane) * 8];
      acc[ns] = __builtin_amdgcn_wmma_f32_16x16x32_bf16(
          false, afr, false, bfr, (short)0, acc[ns], false, false);
    }
    __syncthreads();
  }

  // epilogue: C layout — VGPR r: M = r + 8*(lane>>4), N = lane&15
  const int myrow = row0 + wave * 16;
#pragma unroll
  for (int ns = 0; ns < 4; ++ns) {
    int col = n0 + ns * 16 + lm;
    if (col >= N) continue;
#pragma unroll
    for (int r = 0; r < 8; ++r) {
      int row = myrow + r + (lhalf << 3);
      if (row >= M) continue;
      float v = acc[ns][r];
      if (flags & F_BIAS)  v += bias[row];
      long long ci = (long long)row * ldc + col;
      if (flags & F_RES)   v += Rb[ci];
      if (flags & F_ACCUM) v += Cb[ci];
      Cb[ci] = v;
    }
  }
}

// ---------------------------------------------------------------------------
// Point pipeline kernels
// ---------------------------------------------------------------------------
__global__ void minmax_k(const float* __restrict__ x, const int* __restrict__ sub_idx,
                         float* __restrict__ amin, float* __restrict__ bmax)
{
  __shared__ float smin[256], smax[256];
  int s = blockIdx.x, tid = threadIdx.x;
  const float* coords = x + NTOT;     // x[0,1,:]
  float mn = 1e30f, mx = -1e30f;
  for (int i = tid; i < MSUB; i += 256) {
    float p = coords[sub_idx[s * MSUB + i]] * 10.0f;
    mn = fminf(mn, p); mx = fmaxf(mx, p);
  }
  smin[tid] = mn; smax[tid] = mx; __syncthreads();
  for (int st = 128; st > 0; st >>= 1) {
    if (tid < st) { smin[tid] = fminf(smin[tid], smin[tid + st]);
                    smax[tid] = fmaxf(smax[tid], smax[tid + st]); }
    __syncthreads();
  }
  if (tid == 0) { amin[s] = smin[0]; bmax[s] = smax[0]; }
}

__global__ void interp_k(const float* __restrict__ x, const int* __restrict__ sub_idx,
                         const float* __restrict__ amin, const float* __restrict__ bmax,
                         int* __restrict__ f0, float* __restrict__ w0f, float* __restrict__ w1f)
{
  int i = blockIdx.x * blockDim.x + threadIdx.x;
  if (i >= NTOT) return;
  int sub = i / MSUB;
  const float* coords = x + NTOT;
  float p = coords[sub_idx[i]] * 10.0f;
  float a = amin[sub];
  float d = fmaxf(bmax[sub] - a, EPSF);
  float pos = (p - a) / d * (float)(SG - 1);
  int i0 = (int)floorf(pos);
  if (i0 < 0) i0 = 0; if (i0 > SG - 2) i0 = SG - 2;
  float w1 = pos - (float)i0;
  f0[i] = (sub << 10) + i0;
  w0f[i] = 1.0f - w1;
  w1f[i] = w1;
}

__global__ void den_scatter_k(const int* __restrict__ f0, const float* __restrict__ w0f,
                              const float* __restrict__ w1f, float* __restrict__ den)
{
  int i = blockIdx.x * blockDim.x + threadIdx.x;
  if (i >= NTOT) return;
  int f = f0[i];
  atomicAdd(&den[f],     w0f[i]);
  atomicAdd(&den[f + 1], w1f[i]);
}

__global__ void num_scatter_k(const float* __restrict__ x, const int* __restrict__ sub_idx,
                              const int* __restrict__ f0, const float* __restrict__ w0f,
                              const float* __restrict__ w1f, float* __restrict__ num)
{
  long long gid = (long long)blockIdx.x * blockDim.x + threadIdx.x;
  if (gid >= (long long)BB * NTOT) return;
  int b = (int)(gid / NTOT), i = (int)(gid % NTOT);
  float v = x[(long long)b * 2 * NTOT + sub_idx[i]];   // x[b,0,:]
  int f = f0[i];
  float* nb = num + (long long)b * GRID_TOT;
  atomicAdd(&nb[f],     v * w0f[i]);
  atomicAdd(&nb[f + 1], v * w1f[i]);
}

__global__ void grid_div_k(float* __restrict__ num, const float* __restrict__ den)
{
  int gid = blockIdx.x * blockDim.x + threadIdx.x;
  if (gid >= BB * GRID_TOT) return;
  int f = gid & (GRID_TOT - 1);
  num[gid] = num[gid] / fmaxf(den[f], EPSF);
}

// ---------------------------------------------------------------------------
// Fixed-matrix initializers (Dirichlet upsample, DFT fwd / inverse bases)
// ---------------------------------------------------------------------------
__global__ void init_wup_k(float* __restrict__ Wup)   // [SG x AL], row k, col j
{
  int idx = blockIdx.x * blockDim.x + threadIdx.x;
  if (idx >= SG * AL) return;
  int k = idx / AL, j = idx % AL;
  int t = j - 2 * k;
  float v;
  if (t == 0) v = 1025.0f / 1024.0f;
  else {
    float th = PI_F * (float)t / 1024.0f;
    v = sinf(512.5f * th) / (1024.0f * sinf(0.5f * th));
  }
  Wup[idx] = v;
}

__global__ void init_df_k(float* __restrict__ DF)     // [AL x 64]: cos | -sin
{
  int idx = blockIdx.x * blockDim.x + threadIdx.x;
  if (idx >= AL * 64) return;
  int s = idx / 64, j = idx % 64;
  int m = (j < MODES_) ? j : (j - MODES_);
  float th = (2.0f * PI_F / (float)AL) * (float)m * (float)s;
  DF[idx] = (j < MODES_) ? cosf(th) : -sinf(th);
}

__global__ void init_g_k(float* __restrict__ G)       // [64 x AL] irfft basis
{
  int idx = blockIdx.x * blockDim.x + threadIdx.x;
  if (idx >= 64 * AL) return;
  int j = idx / AL, s = idx % AL;
  int m = (j < MODES_) ? j : (j - MODES_);
  float th = (2.0f * PI_F / (float)AL) * (float)m * (float)s;
  float v;
  if (j < MODES_) v = ((j == 0) ? 1.0f : 2.0f) * (1.0f / (float)AL) * cosf(th);
  else            v = -2.0f * (1.0f / (float)AL) * sinf(th);
  G[idx] = v;
}

// W0 = lift_w @ P_w (64x16), b0 = lift_w @ P_b + lift_b
__global__ void combine_pl_k(const float* __restrict__ P_w, const float* __restrict__ P_b,
                             const float* __restrict__ lift_w, const float* __restrict__ lift_b,
                             float* __restrict__ W0, float* __restrict__ b0)
{
  int tid = threadIdx.x;
  for (int idx = tid; idx < W_ * NSUB; idx += blockDim.x) {
    int o = idx / NSUB, c = idx % NSUB;
    float s = 0.0f;
    for (int k = 0; k < W_; ++k) s += lift_w[o * W_ + k] * P_w[k * NSUB + c];
    W0[idx] = s;
  }
  if (tid < W_) {
    float s = lift_b[tid];
    for (int k = 0; k < W_; ++k) s += lift_w[tid * W_ + k] * P_b[k];
    b0[tid] = s;
  }
}

// Lmix = (Cr + iCi) * (wr + iwi), stored [lr | li] per channel
__global__ void spec_mix_k(const float* __restrict__ F, const float* __restrict__ wr,
                           const float* __restrict__ wi, float* __restrict__ L)
{
  int idx = blockIdx.x * blockDim.x + threadIdx.x;
  if (idx >= BB * W_ * MODES_) return;
  int m = idx % MODES_;
  int c = (idx / MODES_) % W_;
  int b = idx / (W_ * MODES_);
  long long base = (long long)b * (W_ * 64) + (long long)c * 64;
  float Cr = F[base + m], Ci = F[base + MODES_ + m];
  float r = wr[c * MODES_ + m], im = wi[c * MODES_ + m];
  L[base + m]          = Cr * r - Ci * im;
  L[base + MODES_ + m] = Cr * im + Ci * r;
}

// downsample y (B,16,2048)->1024 grid, gather to points, scatter to out
__global__ void gather_out_k(const float* __restrict__ y, const int* __restrict__ f0,
                             const float* __restrict__ w0f, const float* __restrict__ w1f,
                             const int* __restrict__ sub_idx, float* __restrict__ out)
{
  long long gid = (long long)blockIdx.x * blockDim.x + threadIdx.x;
  if (gid >= (long long)BB * NTOT) return;
  int b = (int)(gid / NTOT), i = (int)(gid % NTOT);
  const float* yb = y + (long long)b * NSUB * AL;
  int f = f0[i];
  float acc = 0.0f;
  float wts[2] = { w0f[i], w1f[i] };
#pragma unroll
  for (int q = 0; q < 2; ++q) {
    int ff = f + q;
    int sub = ff >> 10, g = ff & (SG - 1);
    float posr = (float)g * (2047.0f / 1023.0f);
    int j0 = (int)floorf(posr); if (j0 > AL - 2) j0 = AL - 2;
    float t = posr - (float)j0;
    const float* yy = yb + sub * AL;
    acc += (yy[j0] * (1.0f - t) + yy[j0 + 1] * t) * wts[q];
  }
  out[(long long)b * NTOT + sub_idx[i]] = acc;
}

// ---------------------------------------------------------------------------
extern "C" void kernel_launch(void* const* d_in, const int* in_sizes, int n_in,
                              void* d_out, int out_size, void* d_ws, size_t ws_size,
                              hipStream_t stream)
{
  (void)in_sizes; (void)n_in; (void)ws_size;
  const float* x       = (const float*)d_in[0];
  const int*   sub_idx = (const int*)  d_in[1];
  const float* P_w     = (const float*)d_in[2];
  const float* P_b     = (const float*)d_in[3];
  const float* lift_w  = (const float*)d_in[4];
  const float* lift_b  = (const float*)d_in[5];
  const float* spec_wr = (const float*)d_in[6];
  const float* spec_wi = (const float*)d_in[7];
  const float* skip_w  = (const float*)d_in[8];
  const float* skip_b  = (const float*)d_in[9];
  const float* mlp1_w  = (const float*)d_in[10];
  const float* mlp1_b  = (const float*)d_in[11];
  const float* mlp2_w  = (const float*)d_in[12];
  const float* mlp2_b  = (const float*)d_in[13];
  const float* proj_w  = (const float*)d_in[14];
  const float* proj_b  = (const float*)d_in[15];
  const float* Q_w     = (const float*)d_in[16];
  const float* Q_b     = (const float*)d_in[17];
  float* out = (float*)d_out;

  // bump allocator over workspace
  char* wsp = (char*)d_ws;
  auto alloc = [&](size_t bytes) -> void* {
    void* p = (void*)wsp; wsp += (bytes + 255) & ~(size_t)255; return p;
  };
  float* amin = (float*)alloc(16 * 4);
  float* bmax = (float*)alloc(16 * 4);
  float* W0   = (float*)alloc(W_ * NSUB * 4);
  float* b0   = (float*)alloc(W_ * 4);
  int*   f0   = (int*)  alloc((size_t)NTOT * 4);
  float* w0f  = (float*)alloc((size_t)NTOT * 4);
  float* w1f  = (float*)alloc((size_t)NTOT * 4);
  float* den  = (float*)alloc((size_t)GRID_TOT * 4);
  float* num  = (float*)alloc((size_t)BB * GRID_TOT * 4);        // -> grid in place
  float* Wup  = (float*)alloc((size_t)SG * AL * 4);
  float* DF   = (float*)alloc((size_t)AL * 64 * 4);
  float* G    = (float*)alloc((size_t)64 * AL * 4);
  float* ga   = (float*)alloc((size_t)BB * NSUB * AL * 4);
  float* Fbuf = (float*)alloc((size_t)BB * W_ * 64 * 4);
  float* Lmix = (float*)alloc((size_t)BB * W_ * 64 * 4);
  float* hA   = (float*)alloc((size_t)BB * W_ * AL * 4);
  float* hB   = (float*)alloc((size_t)BB * W_ * AL * 4);
  float* u    = (float*)alloc((size_t)BB * W_ * AL * 4);
  float* y    = (float*)alloc((size_t)BB * NSUB * AL * 4);

  const long long sH = (long long)W_ * AL;      // per-batch activation stride
  const long long sF = (long long)W_ * 64;

  auto gemm = [&](const float* A, const float* B, float* C,
                  const float* bias, const float* Res,
                  int M, int N, int K, int lda, int ldb, int ldc,
                  long long sA, long long sB, long long sC, long long sRes,
                  int flags, int batches, int mtiles) {
    dim3 grid((N + 63) / 64, batches, (M + mtiles * 16 - 1) / (mtiles * 16));
    dim3 block(32 * mtiles);
    gemm_wmma_bf16<<<grid, block, 0, stream>>>(A, B, C, bias, Res,
        M, N, K, lda, ldb, ldc, sA, sB, sC, sRes, flags);
  };

  // ---- point scatter onto uniform grids ----
  hipMemsetAsync(den, 0, (size_t)GRID_TOT * 4, stream);
  hipMemsetAsync(num, 0, (size_t)BB * GRID_TOT * 4, stream);
  hipMemsetAsync(out, 0, (size_t)out_size * 4, stream);

  minmax_k<<<NSUB, 256, 0, stream>>>(x, sub_idx, amin, bmax);
  interp_k<<<NTOT / 256, 256, 0, stream>>>(x, sub_idx, amin, bmax, f0, w0f, w1f);
  den_scatter_k<<<NTOT / 256, 256, 0, stream>>>(f0, w0f, w1f, den);
  num_scatter_k<<<(BB * NTOT) / 256, 256, 0, stream>>>(x, sub_idx, f0, w0f, w1f, num);
  grid_div_k<<<(BB * GRID_TOT) / 256, 256, 0, stream>>>(num, den);

  // ---- fixed matrices ----
  init_wup_k<<<(SG * AL) / 256, 256, 0, stream>>>(Wup);
  init_df_k<<<(AL * 64) / 256, 256, 0, stream>>>(DF);
  init_g_k<<<(64 * AL) / 256, 256, 0, stream>>>(G);
  combine_pl_k<<<1, 256, 0, stream>>>(P_w, P_b, lift_w, lift_b, W0, b0);

  // ---- band-limited upsample as GEMM: ga(512,2048) = grid(512,1024) x Wup ----
  gemm(num, Wup, ga, nullptr, nullptr,
       BB * NSUB, AL, SG, SG, AL, AL, 0, 0, 0, 0, 0, /*batches*/1, /*mtiles*/4);

  // ---- fused P+lift: hA = W0(64x16) x ga_b + b0 ----
  gemm(W0, ga, hA, b0, nullptr,
       W_, AL, NSUB, NSUB, AL, AL,
       0, (long long)NSUB * AL, sH, 0, F_BIAS, BB, 4);

  // ---- FNO layers ----
  for (int l = 0; l < NLAYERS; ++l) {
    // F = gelu(hA) x DF    (M=64, K=2048, N=64) : forward truncated DFT
    gemm(hA, DF, Fbuf, nullptr, nullptr,
         W_, 64, AL, AL, 64, 64, sH, 0, sF, 0, F_GELU_A, BB, 4);
    // complex per-channel mode mixing
    spec_mix_k<<<(BB * W_ * MODES_) / 256, 256, 0, stream>>>(
        Fbuf, spec_wr + (size_t)l * W_ * MODES_, spec_wi + (size_t)l * W_ * MODES_, Lmix);
    // hB = skip_w x gelu(hA) + skip_b + hA
    gemm(skip_w + (size_t)l * W_ * W_, hA, hB,
         skip_b + (size_t)l * W_, hA,
         W_, AL, W_, W_, AL, AL, 0, sH, sH, sH,
         F_GELU_B | F_BIAS | F_RES, BB, 4);
    // hB += Lmix x G       (inverse truncated DFT)
    gemm(Lmix, G, hB, nullptr, nullptr,
         W_, AL, 64, 64, AL, AL, sF, 0, sH, 0, F_ACCUM, BB, 4);
    // u = mlp1 x gelu(hB) + b1
    gemm(mlp1_w + (size_t)l * W_ * W_, hB, u,
         mlp1_b + (size_t)l * W_, nullptr,
         W_, AL, W_, W_, AL, AL, 0, sH, sH, 0,
         F_GELU_B | F_BIAS, BB, 4);
    // hA = mlp2 x gelu(u) + b2 + hB
    gemm(mlp2_w + (size_t)l * W_ * W_, u, hA,
         mlp2_b + (size_t)l * W_, hB,
         W_, AL, W_, W_, AL, AL, 0, sH, sH, sH,
         F_GELU_B | F_BIAS | F_RES, BB, 4);
  }

  // ---- proj then Q ----
  gemm(proj_w, hA, hB, proj_b, nullptr,
       W_, AL, W_, W_, AL, AL, 0, sH, sH, 0, F_BIAS, BB, 4);
  gemm(Q_w, hB, y, Q_b, nullptr,
       NSUB, AL, W_, W_, AL, AL, 0, sH, (long long)NSUB * AL, 0, F_BIAS, BB, 1);

  // ---- downsample + gather back to points ----
  gather_out_k<<<(BB * NTOT) / 256, 256, 0, stream>>>(y, f0, w0f, w1f, sub_idx, out);
}
